// AscendQwen3MoeSparseMoeBlock_24885040513652
// MI455X (gfx1250) — compile-verified, matrix operand order checked by reading, not run on hardware
//
#include <hip/hip_runtime.h>
#include <hip/hip_bf16.h>
#include <stdint.h>

// Problem constants (match reference)
#define T_TOK 2048
#define H_DIM 2048
#define E_EXP 32
#define F_DIM 768
#define K_TOP 8

// Tiling
#define MT 32   // tokens per block
#define NT 64   // output cols per pass (4 waves x 16)
#define KT 32   // WMMA K for bf16

typedef __bf16 v16bf __attribute__((ext_vector_type(16)));
typedef __bf16 v8bf  __attribute__((ext_vector_type(8)));
typedef float  v8f   __attribute__((ext_vector_type(8)));
typedef unsigned int u32x4 __attribute__((ext_vector_type(4)));
typedef int          i32x4 __attribute__((ext_vector_type(4)));
typedef int          i32x8 __attribute__((ext_vector_type(8)));

// round-to-nearest-even f32 -> bf16, two at a time, packed into a dword
__device__ __forceinline__ unsigned pack_bf16x2(float a, float b) {
  unsigned ua = __float_as_uint(a);
  unsigned ub = __float_as_uint(b);
  ua += 0x7fffu + ((ua >> 16) & 1u);
  ub += 0x7fffu + ((ub >> 16) & 1u);
  return (ua >> 16) | (ub & 0xffff0000u);
}

__device__ __forceinline__ unsigned short f32_to_bf16_bits(float a) {
  unsigned ua = __float_as_uint(a);
  ua += 0x7fffu + ((ua >> 16) & 1u);
  return (unsigned short)(ua >> 16);
}

// LDS byte offset of a __shared__ object: generic pointers to LDS carry the
// aperture in the high 32 bits; addr[31:0] is the in-LDS offset (ISA 10.2).
__device__ __forceinline__ unsigned lds_offset_of(const void* p) {
  return (unsigned)(uintptr_t)p;
}

// Issue one TDM 2D tile load (bf16 elements) global -> LDS.
// Tile: tile_rows rows of tile_k contiguous elements; row stride in elements.
// D# packing per CDNA5 ISA 8.3/8.4 (group0: count/lds/global/type,
// group1: data_size=2B, tensor dims, tile dims, dim0 stride).
__device__ __forceinline__ void tdm_load_2d_bf16(unsigned lds_off, const void* gtile,
                                                 unsigned tile_k, unsigned tile_rows,
                                                 unsigned tensor_k, unsigned tensor_rows,
                                                 unsigned row_stride_elems) {
  uint64_t ga = (uint64_t)(uintptr_t)gtile;
  u32x4 g0;
  g0[0] = 1u;                                                 // count=1 (valid user D#)
  g0[1] = lds_off;                                            // lds_addr [63:32]
  g0[2] = (unsigned)ga;                                       // global_addr [95:64]
  g0[3] = (unsigned)((ga >> 32) & 0x01ffffffu) | (2u << 30);  // addr[56:32] | type=2
  i32x8 g1;
  g1[0] = 1 << 16;                                            // data_size=1 (2 bytes)
  g1[1] = (int)((tensor_k & 0xffffu) << 16);                  // tensor_dim0[15:0]
  g1[2] = (int)(((tensor_k >> 16) & 0xffffu) |
                ((tensor_rows & 0xffffu) << 16));             // dim0[31:16] | dim1[15:0]
  g1[3] = (int)(((tensor_rows >> 16) & 0xffffu) |
                ((tile_k & 0xffffu) << 16));                  // dim1[31:16] | tile_dim0
  g1[4] = (int)(tile_rows & 0xffffu);                         // tile_dim1 (tile_dim2=0)
  g1[5] = (int)row_stride_elems;                              // tensor_dim0_stride[31:0]
  g1[6] = 0;                                                  // stride hi / dim1_stride
  g1[7] = 0;
  i32x4 z4 = {0, 0, 0, 0};
#if defined(__clang_major__) && (__clang_major__ >= 23)
  i32x8 z8 = {0, 0, 0, 0, 0, 0, 0, 0};
  __builtin_amdgcn_tensor_load_to_lds(g0, g1, z4, z4, z8, 0);
#else
  __builtin_amdgcn_tensor_load_to_lds(g0, g1, z4, z4, 0);
#endif
}

// Load a 16x32 bf16 A/B fragment for v_wmma_f32_16x16x32_bf16 from a
// row-major LDS region (row = M for A, row = N for transposed-B).
__device__ __forceinline__ v16bf load_frag16x32(const __bf16* base, int stride,
                                                int row, int kofs, int lane) {
  const __bf16* p = base + row * stride + kofs + ((lane & 16) ? 8 : 0);
  v8bf lo = *(const v8bf*)(p);
  v8bf hi = *(const v8bf*)(p + 16);
  v16bf r;
#pragma unroll
  for (int i = 0; i < 8; ++i) { r[i] = lo[i]; r[i + 8] = hi[i]; }
  return r;
}

// ---------------------------------------------------------------------------
// Router: logits -> softmax -> top-8 -> renormalize -> dense route[T, 32]
// ---------------------------------------------------------------------------
__global__ __launch_bounds__(32)
void moe_router_kernel(const float* __restrict__ x,
                       const float* __restrict__ gw,
                       float* __restrict__ route) {
  const int t = blockIdx.x;
  const int e = threadIdx.x;
  const float* xr = x + (size_t)t * H_DIM;

  float logit = 0.f;
#pragma unroll 4
  for (int h = 0; h < H_DIM; ++h)
    logit = fmaf(xr[h], gw[h * E_EXP + e], logit);

  float m = logit;
#pragma unroll
  for (int off = 16; off; off >>= 1) m = fmaxf(m, __shfl_xor(m, off));
  float p = __expf(logit - m);
  float s = p;
#pragma unroll
  for (int off = 16; off; off >>= 1) s += __shfl_xor(s, off);
  float prob = p / s;

  float work = prob, picked = 0.f;
#pragma unroll
  for (int i = 0; i < K_TOP; ++i) {
    float mx = work;
#pragma unroll
    for (int off = 16; off; off >>= 1) mx = fmaxf(mx, __shfl_xor(mx, off));
    unsigned long long msk = __ballot(work == mx);
    int lead = __ffsll(msk) - 1;
    if (e == lead) { picked = prob; work = -1.f; }
  }
  float ps = picked;
#pragma unroll
  for (int off = 16; off; off >>= 1) ps += __shfl_xor(ps, off);
  route[t * E_EXP + e] = picked / ps;
}

// ---------------------------------------------------------------------------
// fp32 -> packed bf16 elementwise (for x)
// ---------------------------------------------------------------------------
__global__ __launch_bounds__(256)
void cvt_bf16_kernel(const float* __restrict__ in, unsigned* __restrict__ outp,
                     int npairs) {
  int i = blockIdx.x * 256 + threadIdx.x;
  if (i < npairs) {
    float2 v = *(const float2*)&in[2 * i];
    outp[i] = pack_bf16x2(v.x, v.y);
  }
}

// ---------------------------------------------------------------------------
// Per-expert transpose + convert: in [E][R][C] fp32 -> out [E][C][R] bf16.
// 32x32 tiles through LDS; bf16 pairs packed along R on the way out.
// ---------------------------------------------------------------------------
__global__ __launch_bounds__(256)
void transpose_cvt_kernel(const float* __restrict__ in, unsigned* __restrict__ outp,
                          int R, int C) {
  __shared__ float t[32][33];
  const int e  = blockIdx.z;
  const int c0 = blockIdx.x * 32;
  const int r0 = blockIdx.y * 32;
  const float* src = in + (size_t)e * R * C;
  unsigned* dst = outp + ((size_t)e * C * R) / 2;
  const int tid = threadIdx.x;
#pragma unroll
  for (int i = 0; i < 4; ++i) {
    int idx = tid + i * 256;
    int rr = idx >> 5, cc = idx & 31;
    t[rr][cc] = src[(size_t)(r0 + rr) * C + c0 + cc];
  }
  __syncthreads();
#pragma unroll
  for (int i = 0; i < 2; ++i) {
    int idx = tid + i * 256;              // 0..511 output dwords
    int cc = idx >> 4, pp = idx & 15;
    dst[((size_t)(c0 + cc) * R + r0) / 2 + pp] =
        pack_bf16x2(t[2 * pp][cc], t[2 * pp + 1][cc]);
  }
}

// ---------------------------------------------------------------------------
// TDM-staged expert FFN. Inputs are pre-converted bf16:
//   xb  [T][H], wgT/wuT [E][F][H] (transposed), wdT [E][H][F] (transposed).
// All tiles DMA'd to LDS by the Tensor Data Mover; hot loop = WMMA only.
// ---------------------------------------------------------------------------
__global__ __launch_bounds__(256)
void moe_expert_kernel_tdm(const __bf16* __restrict__ xb,
                           const __bf16* __restrict__ wgT,
                           const __bf16* __restrict__ wuT,
                           const __bf16* __restrict__ wdT,
                           const float* __restrict__ route,
                           float* __restrict__ out) {
  __shared__ __align__(16) __bf16 xs[MT][KT];     // 2 KB
  __shared__ __align__(16) __bf16 wtA[NT][KT];    // 4 KB
  __shared__ __align__(16) __bf16 wtB[NT][KT];    // 4 KB
  __shared__ __align__(16) __bf16 hs[MT][F_DIM];  // 48 KB
  __shared__ float rt[MT];

  const int tid  = threadIdx.x;
  const int lane = tid & 31;
  const int wave = tid >> 5;
  const int wm   = wave >> 2;
  const int wn   = wave & 3;
  const int t0   = blockIdx.x * MT;
  const int e    = blockIdx.y;

  const __bf16* wgT_e = wgT + (size_t)e * F_DIM * H_DIM;
  const __bf16* wuT_e = wuT + (size_t)e * F_DIM * H_DIM;
  const __bf16* wdT_e = wdT + (size_t)e * H_DIM * F_DIM;

  if (tid < MT) rt[tid] = route[(size_t)(t0 + tid) * E_EXP + e];

  const unsigned xs_off = lds_offset_of(&xs[0][0]);
  const unsigned wa_off = lds_offset_of(&wtA[0][0]);
  const unsigned wb_off = lds_offset_of(&wtB[0][0]);

  const int arow = wm * 16 + (lane & 15);
  const int brow = wn * 16 + (lane & 15);

  // -------- phase 1: h = silu(x@wg) * (x@wu) --------
  for (int nb = 0; nb < F_DIM; nb += NT) {
    v8f accG = {};
    v8f accU = {};
    for (int k0 = 0; k0 < H_DIM; k0 += KT) {
      __syncthreads();                 // previous tile consumers done
      if (wave == 0) {
        tdm_load_2d_bf16(xs_off, xb + (size_t)t0 * H_DIM + k0,
                         KT, MT, H_DIM, T_TOK, H_DIM);
        tdm_load_2d_bf16(wa_off, wgT_e + (size_t)nb * H_DIM + k0,
                         KT, NT, H_DIM, F_DIM, H_DIM);
        tdm_load_2d_bf16(wb_off, wuT_e + (size_t)nb * H_DIM + k0,
                         KT, NT, H_DIM, F_DIM, H_DIM);
        __builtin_amdgcn_s_wait_tensorcnt(0);
      }
      __syncthreads();                 // tiles visible to all waves
      v16bf a  = load_frag16x32(&xs[0][0],  KT, arow, 0, lane);
      v16bf bg = load_frag16x32(&wtA[0][0], KT, brow, 0, lane);
      v16bf bu = load_frag16x32(&wtB[0][0], KT, brow, 0, lane);
      accG = __builtin_amdgcn_wmma_f32_16x16x32_bf16(false, a, false, bg,
                                                     (short)0, accG, false, false);
      accU = __builtin_amdgcn_wmma_f32_16x16x32_bf16(false, a, false, bu,
                                                     (short)0, accU, false, false);
    }
#pragma unroll
    for (int j = 0; j < 8; ++j) {
      int m = wm * 16 + j + ((lane & 16) ? 8 : 0);
      int n = nb + wn * 16 + (lane & 15);
      float g = accG[j];
      float sig = __builtin_amdgcn_rcpf(1.f + __expf(-g));
      *(unsigned short*)&hs[m][n] = f32_to_bf16_bits(g * sig * accU[j]);
    }
  }

  // -------- phase 2: out += route * (h @ wd) --------
  for (int hb = 0; hb < H_DIM; hb += NT) {
    v8f acc = {};
    for (int k0 = 0; k0 < F_DIM; k0 += KT) {
      __syncthreads();
      if (wave == 0) {
        tdm_load_2d_bf16(wa_off, wdT_e + (size_t)hb * F_DIM + k0,
                         KT, NT, F_DIM, H_DIM, F_DIM);
        __builtin_amdgcn_s_wait_tensorcnt(0);
      }
      __syncthreads();
      v16bf a = load_frag16x32(&hs[0][0],  F_DIM, arow, k0, lane);
      v16bf b = load_frag16x32(&wtA[0][0], KT,    brow, 0,  lane);
      acc = __builtin_amdgcn_wmma_f32_16x16x32_bf16(false, a, false, b,
                                                    (short)0, acc, false, false);
    }
#pragma unroll
    for (int j = 0; j < 8; ++j) {
      int m = wm * 16 + j + ((lane & 16) ? 8 : 0);
      int n = hb + wn * 16 + (lane & 15);
      atomicAdd(&out[(size_t)(t0 + m) * H_DIM + n], acc[j] * rt[m]);
    }
  }
}

// ---------------------------------------------------------------------------
// Fallback expert FFN (no workspace for bf16 weights): VALU-staged tiles.
// ---------------------------------------------------------------------------
__global__ __launch_bounds__(256)
void moe_expert_kernel(const float* __restrict__ x,
                       const float* __restrict__ wg,
                       const float* __restrict__ wu,
                       const float* __restrict__ wd,
                       const float* __restrict__ route,
                       float* __restrict__ out) {
  __shared__ __align__(16) __bf16 xs[MT][KT];
  __shared__ __align__(16) __bf16 wtA[NT][KT];
  __shared__ __align__(16) __bf16 wtB[NT][KT];
  __shared__ __align__(16) __bf16 hs[MT][F_DIM];
  __shared__ float rt[MT];

  const int tid  = threadIdx.x;
  const int lane = tid & 31;
  const int wave = tid >> 5;
  const int wm   = wave >> 2;
  const int wn   = wave & 3;
  const int t0   = blockIdx.x * MT;
  const int e    = blockIdx.y;

  const float* wg_e = wg + (size_t)e * H_DIM * F_DIM;
  const float* wu_e = wu + (size_t)e * H_DIM * F_DIM;
  const float* wd_e = wd + (size_t)e * F_DIM * H_DIM;

  if (tid < MT) rt[tid] = route[(size_t)(t0 + tid) * E_EXP + e];

  const int arow = wm * 16 + (lane & 15);
  const int brow = wn * 16 + (lane & 15);
  const int wn_c = tid & (NT - 1);
  const int wk_p = tid >> 6;

  for (int nb = 0; nb < F_DIM; nb += NT) {
    v8f accG = {};
    v8f accU = {};
    for (int k0 = 0; k0 < H_DIM; k0 += KT) {
      __syncthreads();
#pragma unroll
      for (int i = 0; i < 2; ++i) {
        int idx = tid + i * 256;
        int r = idx >> 4, k = (idx & 15) * 2;
        float2 v = *(const float2*)&x[(size_t)(t0 + r) * H_DIM + k0 + k];
        *(unsigned*)&xs[r][k] = pack_bf16x2(v.x, v.y);
      }
#pragma unroll
      for (int i = 0; i < 4; ++i) {
        int k = (wk_p + i * 4) * 2;
        size_t g0 = (size_t)(k0 + k) * F_DIM + nb + wn_c;
        *(unsigned*)&wtA[wn_c][k] = pack_bf16x2(wg_e[g0], wg_e[g0 + F_DIM]);
        *(unsigned*)&wtB[wn_c][k] = pack_bf16x2(wu_e[g0], wu_e[g0 + F_DIM]);
      }
      if (k0 + KT < H_DIM) {
        size_t gn = (size_t)(k0 + KT + wk_p * 2) * F_DIM + nb + wn_c;
        __builtin_prefetch(&wg_e[gn], 0, 0);
        __builtin_prefetch(&wu_e[gn], 0, 0);
      }
      __syncthreads();
      v16bf a  = load_frag16x32(&xs[0][0],  KT, arow, 0, lane);
      v16bf bg = load_frag16x32(&wtA[0][0], KT, brow, 0, lane);
      v16bf bu = load_frag16x32(&wtB[0][0], KT, brow, 0, lane);
      accG = __builtin_amdgcn_wmma_f32_16x16x32_bf16(false, a, false, bg,
                                                     (short)0, accG, false, false);
      accU = __builtin_amdgcn_wmma_f32_16x16x32_bf16(false, a, false, bu,
                                                     (short)0, accU, false, false);
    }
#pragma unroll
    for (int j = 0; j < 8; ++j) {
      int m = wm * 16 + j + ((lane & 16) ? 8 : 0);
      int n = nb + wn * 16 + (lane & 15);
      float g = accG[j];
      float sig = __builtin_amdgcn_rcpf(1.f + __expf(-g));
      *(unsigned short*)&hs[m][n] = f32_to_bf16_bits(g * sig * accU[j]);
    }
  }

  for (int hb = 0; hb < H_DIM; hb += NT) {
    v8f acc = {};
    for (int k0 = 0; k0 < F_DIM; k0 += KT) {
      __syncthreads();
#pragma unroll
      for (int i = 0; i < 4; ++i) {
        int k = (wk_p + i * 4) * 2;
        size_t g0 = (size_t)(k0 + k) * H_DIM + hb + wn_c;
        *(unsigned*)&wtA[wn_c][k] = pack_bf16x2(wd_e[g0], wd_e[g0 + H_DIM]);
      }
      if (k0 + KT < F_DIM) {
        size_t gn = (size_t)(k0 + KT + wk_p * 2) * H_DIM + hb + wn_c;
        __builtin_prefetch(&wd_e[gn], 0, 0);
      }
      __syncthreads();
      v16bf a = load_frag16x32(&hs[0][0],  F_DIM, arow, k0, lane);
      v16bf b = load_frag16x32(&wtA[0][0], KT,    brow, 0,  lane);
      acc = __builtin_amdgcn_wmma_f32_16x16x32_bf16(false, a, false, b,
                                                    (short)0, acc, false, false);
    }
#pragma unroll
    for (int j = 0; j < 8; ++j) {
      int m = wm * 16 + j + ((lane & 16) ? 8 : 0);
      int n = hb + wn * 16 + (lane & 15);
      atomicAdd(&out[(size_t)(t0 + m) * H_DIM + n], acc[j] * rt[m]);
    }
  }
}

// ---------------------------------------------------------------------------
extern "C" void kernel_launch(void* const* d_in, const int* in_sizes, int n_in,
                              void* d_out, int out_size, void* d_ws, size_t ws_size,
                              hipStream_t stream) {
  const float* x      = (const float*)d_in[0];  // [T, H]
  const float* gate_w = (const float*)d_in[1];  // [H, E]
  const float* wgp    = (const float*)d_in[2];  // [E, H, F]
  const float* wup    = (const float*)d_in[3];  // [E, H, F]
  const float* wdp    = (const float*)d_in[4];  // [E, F, H]
  float* out          = (float*)d_out;          // [T, H]

  // workspace layout
  const size_t route_bytes = (size_t)T_TOK * E_EXP * sizeof(float);
  const size_t xb_bytes    = (size_t)T_TOK * H_DIM * 2;
  const size_t w_bytes     = (size_t)E_EXP * F_DIM * H_DIM * 2;
  size_t off = 0;
  char* ws = (char*)d_ws;
  float* route = (float*)(ws + off); off += route_bytes;
  const size_t xb_off = off; off += xb_bytes;
  const size_t wg_off = off; off += w_bytes;
  const size_t wu_off = off; off += w_bytes;
  const size_t wd_off = off; off += w_bytes;
  const bool use_tdm = (ws_size >= off);   // constant per harness -> deterministic

  hipMemsetAsync(d_out, 0, (size_t)T_TOK * H_DIM * sizeof(float), stream);

  moe_router_kernel<<<T_TOK, 32, 0, stream>>>(x, gate_w, route);

  dim3 grid(T_TOK / MT, E_EXP);
  if (use_tdm) {
    // one-time bf16 conversion (+ transpose for weights) into workspace
    int xpairs = T_TOK * H_DIM / 2;
    cvt_bf16_kernel<<<(xpairs + 255) / 256, 256, 0, stream>>>(
        x, (unsigned*)(ws + xb_off), xpairs);
    transpose_cvt_kernel<<<dim3(F_DIM / 32, H_DIM / 32, E_EXP), 256, 0, stream>>>(
        wgp, (unsigned*)(ws + wg_off), H_DIM, F_DIM);
    transpose_cvt_kernel<<<dim3(F_DIM / 32, H_DIM / 32, E_EXP), 256, 0, stream>>>(
        wup, (unsigned*)(ws + wu_off), H_DIM, F_DIM);
    transpose_cvt_kernel<<<dim3(H_DIM / 32, F_DIM / 32, E_EXP), 256, 0, stream>>>(
        wdp, (unsigned*)(ws + wd_off), F_DIM, H_DIM);

    moe_expert_kernel_tdm<<<grid, 256, 0, stream>>>(
        (const __bf16*)(ws + xb_off), (const __bf16*)(ws + wg_off),
        (const __bf16*)(ws + wu_off), (const __bf16*)(ws + wd_off), route, out);
  } else {
    moe_expert_kernel<<<grid, 256, 0, stream>>>(x, wgp, wup, wdp, route, out);
  }
}
